// MappedTSTEncoderLayerShared_53334903881766
// MI455X (gfx1250) — matrix operand order repaired and checked by reading, not compiled, not run on hardware
//
#include <hip/hip_runtime.h>
#include <hip/hip_bf16.h>
#include <math.h>

typedef __bf16 bf16;
typedef __attribute__((ext_vector_type(16))) __bf16 v16bf;
typedef __attribute__((ext_vector_type(8)))  __bf16 v8bf;
typedef __attribute__((ext_vector_type(8)))  float  v8f;

// ---------- helpers ----------
__device__ __forceinline__ bf16 f2bf(float f) {
    union { float f; unsigned u; } v; v.f = f;
    unsigned r = v.u + 0x7FFFu + ((v.u >> 16) & 1u);   // round-to-nearest-even
    unsigned short h = (unsigned short)(r >> 16);
    return __builtin_bit_cast(bf16, h);
}
__device__ __forceinline__ bf16 bf0() {
    return __builtin_bit_cast(bf16, (unsigned short)0);
}
__device__ __forceinline__ void store_c(float* p, float v) { *p = v; }
__device__ __forceinline__ void store_c(bf16*  p, float v) { *p = f2bf(v); }

// ---------- CDNA5 async global->LDS copies (ASYNCcnt path) ----------
// LDS byte offset = low 32 bits of the flat shared-aperture address (ISA 10.2).
__device__ __forceinline__ void async_b128(void* lds, const void* g) {
    asm volatile("global_load_async_to_lds_b128 %0, %1, off"
                 :: "v"((unsigned)(unsigned long long)(uintptr_t)lds),
                    "v"((unsigned long long)(uintptr_t)g)
                 : "memory");
}
__device__ __forceinline__ void async_b64(void* lds, const void* g) {
    asm volatile("global_load_async_to_lds_b64 %0, %1, off"
                 :: "v"((unsigned)(unsigned long long)(uintptr_t)lds),
                    "v"((unsigned long long)(uintptr_t)g)
                 : "memory");
}
__device__ __forceinline__ void wait_async() {
    asm volatile("s_wait_asynccnt 0x0" ::: "memory");
}

// 256-thread block sum (8 waves of 32)
__device__ __forceinline__ float block_sum(float v, float* red) {
    #pragma unroll
    for (int o = 16; o > 0; o >>= 1) v += __shfl_xor(v, o, 32);
    int wv = threadIdx.x >> 5;
    if ((threadIdx.x & 31) == 0) red[wv] = v;
    __syncthreads();
    float s = 0.0f;
    #pragma unroll
    for (int i = 0; i < 8; ++i) s += red[i];
    __syncthreads();
    return s;
}

// fp32 -> bf16 array conversion (weights, done once)
__global__ __launch_bounds__(256) void cvt_bf16_kernel(
    const float* __restrict__ x, bf16* __restrict__ y, int n)
{
    int i = blockIdx.x * 256 + threadIdx.x;
    if (i < n) y[i] = f2bf(x[i]);
}

// =====================================================================
// Generic GEMM:  C[M,N] = act(A[M,K] @ W[N,K]^T + bias[N]),  W is bf16.
// BM=128 BN=128 BK=32, 8 waves (2x4), wave tile 64x32 = 4x2 WMMA tiles.
// bf16 A tiles + all W tiles copied with global_load_async_to_lds_b128.
// ACT: 0=none, 1=exact GELU.
// =====================================================================
template<typename AT, typename CT, int ACT>
__global__ __launch_bounds__(256) void gemm_kernel(
    const AT* __restrict__ A, const bf16* __restrict__ W,
    const float* __restrict__ bias, CT* __restrict__ C,
    int M, int N, int K)
{
    constexpr int BM = 128, BN = 128, BK = 32;
    __shared__ alignas(16) bf16 As[2][BM][BK + 8];
    __shared__ alignas(16) bf16 Bs[2][BN][BK + 8];

    const int tid  = threadIdx.x;
    const int lane = tid & 31;
    const int wv   = tid >> 5;
    const int wm   = wv >> 2;          // 0..1
    const int wn   = wv & 3;           // 0..3
    const int m0   = blockIdx.y * BM;
    const int n0   = blockIdx.x * BN;
    const int nk   = K / BK;

    v8f acc[4][2];
    #pragma unroll
    for (int i = 0; i < 4; ++i)
        #pragma unroll
        for (int j = 0; j < 2; ++j) acc[i][j] = {};

    auto load_stage = [&](int kk, int buf) {
        const int k0 = kk * BK;
        // W tile (bf16): 128 rows x 64B, async 16B per lane, 2 per thread
        #pragma unroll
        for (int i = 0; i < 2; ++i) {
            int idx = i * 256 + tid;
            int r = idx >> 2, seg = idx & 3;
            async_b128(&Bs[buf][r][seg * 8],
                       W + (size_t)(n0 + r) * K + k0 + seg * 8);
        }
        if constexpr (__is_same(AT, bf16)) {
            #pragma unroll
            for (int i = 0; i < 2; ++i) {
                int idx = i * 256 + tid;
                int r = idx >> 2, seg = idx & 3;
                async_b128(&As[buf][r][seg * 8],
                           A + (size_t)(m0 + r) * K + k0 + seg * 8);
            }
        } else {
            #pragma unroll
            for (int i = 0; i < 16; ++i) {       // fp32 A: convert via VALU
                int idx = i * 256 + tid;
                int r = idx >> 5, c = idx & 31;
                As[buf][r][c] = f2bf(A[(size_t)(m0 + r) * K + k0 + c]);
            }
        }
    };
    // A frag (16x32): lane row = lane%16; lanes0-15 K:0-7,16-23; lanes16-31 K:8-15,24-31
    auto frag_a = [&](int buf, int mt) -> v16bf {
        const bf16* row = &As[buf][64 * wm + 16 * mt + (lane & 15)][0];
        int kb = (lane & 16) ? 8 : 0;
        v8bf lo = *(const v8bf*)(row + kb);
        v8bf hi = *(const v8bf*)(row + 16 + kb);
        v16bf r;
        #pragma unroll
        for (int i = 0; i < 8; ++i) { r[i] = lo[i]; r[8 + i] = hi[i]; }
        return r;
    };
    // B frag (32x16): lane col = lane%16; lanes0-15 K:0-7,16-23; lanes16-31 K:8-15,24-31 (read W rows = B^T)
    auto frag_b = [&](int buf, int nt) -> v16bf {
        const bf16* row = &Bs[buf][32 * wn + 16 * nt + (lane & 15)][0];
        int kb = (lane & 16) ? 8 : 0;
        v8bf lo = *(const v8bf*)(row + kb);
        v8bf hi = *(const v8bf*)(row + 16 + kb);
        v16bf r;
        #pragma unroll
        for (int i = 0; i < 8; ++i) { r[i] = lo[i]; r[8 + i] = hi[i]; }
        return r;
    };

    load_stage(0, 0);
    wait_async();
    __syncthreads();
    for (int kk = 0; kk < nk; ++kk) {
        const int buf = kk & 1;
        if (kk + 1 < nk) load_stage(kk + 1, buf ^ 1);
        v16bf bfr0 = frag_b(buf, 0);
        v16bf bfr1 = frag_b(buf, 1);
        #pragma unroll
        for (int i = 0; i < 4; ++i) {
            v16bf af = frag_a(buf, i);
            acc[i][0] = __builtin_amdgcn_wmma_f32_16x16x32_bf16(
                false, af, false, bfr0, (short)0, acc[i][0], false, false);
            acc[i][1] = __builtin_amdgcn_wmma_f32_16x16x32_bf16(
                false, af, false, bfr1, (short)0, acc[i][1], false, false);
        }
        wait_async();
        __syncthreads();
    }

    // epilogue: C tile layout: vgpr r -> row r + 8*(lane/16), col = lane%16
    const int roff = (lane >> 4) << 3;
    const int cl   = lane & 15;
    #pragma unroll
    for (int i = 0; i < 4; ++i) {
        const int gr0 = m0 + 64 * wm + 16 * i;
        #pragma unroll
        for (int j = 0; j < 2; ++j) {
            const int gc = n0 + 32 * wn + 16 * j + cl;
            const float bv = bias[gc];
            #pragma unroll
            for (int r = 0; r < 8; ++r) {
                float v = acc[i][j][r] + bv;
                if (ACT == 1) v = 0.5f * v * (1.0f + erff(v * 0.70710678118f));
                store_c(&C[(size_t)(gr0 + r + roff) * N + gc], v);
            }
        }
    }
}

// =====================================================================
// Flash attention per (b,h): Q = router[h] [128,16] (K padded to 32),
// K=V = kbf[b,:,h,:] [4096,16] (bf16). Online softmax, S chunks of 64.
// K chunks copied global->LDS with async b64 ops.
// =====================================================================
__global__ __launch_bounds__(256) void attn_kernel(
    const bf16* __restrict__ kbf,      // [16,4096,256] bf16
    const float* __restrict__ router,  // [16,128,16]
    float* __restrict__ ctx)           // [16,128,256]
{
    constexpr int SC = 64;
    __shared__ alignas(16) bf16  Ks[SC][16];
    __shared__ float Ss[8][16][SC + 4];
    __shared__ float rowtmp[8][16];

    const int tid  = threadIdx.x, lane = tid & 31, wv = tid >> 5;
    const int h = blockIdx.x, b = blockIdx.y;
    const int row16 = lane & 15;
    const int part  = lane >> 4;
    const int roff  = part << 3;

    // Q fragment, pre-scaled by dk^-0.5 = 0.25; K dim 16..31 zero-padded
    v16bf qf;
    {
        const float* qr = router + ((size_t)h * 128 + 16 * wv + row16) * 16;
        #pragma unroll
        for (int e = 0; e < 16; ++e) {
            int v = e >> 1;
            if (part == 0) {
                int k = e;
                qf[e] = f2bf(qr[k] * 0.25f);
            } else qf[e] = bf0();
        }
    }

    v8f O = {};
    float mrun = -INFINITY, lrun = 0.0f;

    for (int t = 0; t < 4096 / SC; ++t) {
        const int s0 = t * SC;
        __syncthreads();
        {   // async copy of K chunk: 64 rows x 32B, 8B per thread
            int s = tid >> 2, p8 = tid & 3;
            async_b64(&Ks[s][p8 * 4],
                      kbf + (size_t)(b * 4096 + s0 + s) * 256 + h * 16 + p8 * 4);
        }
        wait_async();
        __syncthreads();

        // scores: 4 WMMAs (B lanes16-31 cover K 16..31 = zero pad)
        #pragma unroll
        for (int j = 0; j < SC / 16; ++j) {
            v16bf bf;
            if (part == 0) {
                const bf16* kr = &Ks[16 * j + row16][0];
                v8bf lo = *(const v8bf*)(kr);
                v8bf hi = *(const v8bf*)(kr + 8);
                #pragma unroll
                for (int i = 0; i < 8; ++i) { bf[i] = lo[i]; bf[8 + i] = hi[i]; }
            } else {
                #pragma unroll
                for (int i = 0; i < 16; ++i) bf[i] = bf0();
            }
            v8f sc = {};
            sc = __builtin_amdgcn_wmma_f32_16x16x32_bf16(
                false, qf, false, bf, (short)0, sc, false, false);
            #pragma unroll
            for (int r = 0; r < 8; ++r)
                Ss[wv][r + roff][16 * j + row16] = sc[r];
        }

        // wave-local online softmax (2 lanes per row)
        float mloc = -INFINITY;
        #pragma unroll 8
        for (int jj = 0; jj < SC / 2; ++jj)
            mloc = fmaxf(mloc, Ss[wv][row16][part * (SC / 2) + jj]);
        mloc = fmaxf(mloc, __shfl_xor(mloc, 16, 32));
        const float mnew = fmaxf(mrun, mloc);
        float sloc = 0.0f;
        #pragma unroll 8
        for (int jj = 0; jj < SC / 2; ++jj) {
            int c = part * (SC / 2) + jj;
            float p = __expf(Ss[wv][row16][c] - mnew);
            Ss[wv][row16][c] = p;
            sloc += p;
        }
        sloc += __shfl_xor(sloc, 16, 32);
        const float fac = __expf(mrun - mnew);
        lrun = lrun * fac + sloc;
        mrun = mnew;
        rowtmp[wv][row16] = fac;
        #pragma unroll
        for (int r = 0; r < 8; ++r) O[r] *= rowtmp[wv][r + roff];

        // P @ V : two K=32 WMMAs over the 64-wide chunk
        #pragma unroll
        for (int half = 0; half < 2; ++half) {
            v16bf pf, vf;
            #pragma unroll
            for (int e = 0; e < 16; ++e) {
                int v = e >> 1;
                int ka = 32 * half + ((v < 4) ? 0 : 16) + (part ? 8 : 0) + 2 * (v & 3) + (e & 1);
                pf[e] = f2bf(Ss[wv][row16][ka]);
                int kb = 32 * half + ((e < 8) ? 0 : 16) + part * 8 + (e & 7);
                vf[e] = Ks[kb][row16];
            }
            O = __builtin_amdgcn_wmma_f32_16x16x32_bf16(
                false, pf, false, vf, (short)0, O, false, false);
        }
    }

    rowtmp[wv][row16] = 1.0f / lrun;
    #pragma unroll
    for (int r = 0; r < 8; ++r) {
        int c = 16 * wv + r + roff;
        ctx[((size_t)b * 128 + c) * 256 + h * 16 + row16] = O[r] * rowtmp[wv][r + roff];
    }
}

// =====================================================================
// expand (m_expand @ (proj+pos)) * scale_tf, + src residual, LayerNorm1
// writes y in fp32 (residual) and bf16 (FFN1 A input)
// =====================================================================
__global__ __launch_bounds__(256) void expand_ln1_kernel(
    const float* __restrict__ src, const float* __restrict__ proj,
    const float* __restrict__ pos, const float* __restrict__ mexp,
    const float* __restrict__ scale_tf, const float* __restrict__ g,
    const float* __restrict__ be, float* __restrict__ y,
    bf16* __restrict__ ybf)
{
    __shared__ float me[128];
    __shared__ float red[8];
    const int tid = threadIdx.x;
    const int q = blockIdx.x, b = blockIdx.y;
    if (tid < 128) me[tid] = mexp[q * 128 + tid];
    __syncthreads();
    float acc = 0.0f;
    for (int c = 0; c < 128; ++c)
        acc = fmaf(me[c], proj[((size_t)b * 128 + c) * 256 + tid] + pos[c * 256 + tid], acc);
    for (int x = 0; x < 8; ++x) {
        size_t base = ((size_t)(b * 8 + x) * 512 + q) * 256;
        float v = src[base + tid] + acc * scale_tf[x];
        float mu  = block_sum(v, red) * (1.0f / 256.0f);
        float d   = v - mu;
        float var = block_sum(d * d, red) * (1.0f / 256.0f);
        float o   = d * rsqrtf(var + 1e-5f) * g[tid] + be[tid];
        y[base + tid]   = o;
        ybf[base + tid] = f2bf(o);
    }
}

// residual add + LayerNorm2 -> output
__global__ __launch_bounds__(256) void add_ln_kernel(
    const float* __restrict__ y, const float* __restrict__ z,
    const float* __restrict__ g, const float* __restrict__ be,
    float* __restrict__ out)
{
    __shared__ float red[8];
    const size_t row = blockIdx.x;
    const int tid = threadIdx.x;
    float v = y[row * 256 + tid] + z[row * 256 + tid];
    float mu  = block_sum(v, red) * (1.0f / 256.0f);
    float d   = v - mu;
    float var = block_sum(d * d, red) * (1.0f / 256.0f);
    out[row * 256 + tid] = d * rsqrtf(var + 1e-5f) * g[tid] + be[tid];
}

// =====================================================================
extern "C" void kernel_launch(void* const* d_in, const int* in_sizes, int n_in,
                              void* d_out, int out_size, void* d_ws, size_t ws_size,
                              hipStream_t stream) {
    const float* src    = (const float*)d_in[0];
    const float* router = (const float*)d_in[1];
    const float* Wk_w   = (const float*)d_in[2];
    const float* Wk_b   = (const float*)d_in[3];
    const float* out_w  = (const float*)d_in[4];
    const float* out_b  = (const float*)d_in[5];
    const float* m_exp  = (const float*)d_in[6];
    const float* pos    = (const float*)d_in[7];
    const float* sc_tf  = (const float*)d_in[8];
    const float* ln1_g  = (const float*)d_in[9];
    const float* ln1_b  = (const float*)d_in[10];
    const float* ff1_w  = (const float*)d_in[11];
    const float* ff1_b  = (const float*)d_in[12];
    const float* ff2_w  = (const float*)d_in[13];
    const float* ff2_b  = (const float*)d_in[14];
    const float* ln2_g  = (const float*)d_in[15];
    const float* ln2_b  = (const float*)d_in[16];
    float* out = (float*)d_out;

    const int BS = 16, DM = 256, NH = 16, DC = 128, DFF = 1024, QL = 512;
    const int Srows = 65536;                       // bs * dsc * qlen

    char* w = (char*)d_ws;
    float* zbuf = (float*)w;            w += (size_t)Srows * DM * 4;     // 64 MiB
    bf16*  kbf  = (bf16*)zbuf;          // alias: k (bf16) dead before FFN2 writes z
    float* ybuf = (float*)w;            w += (size_t)Srows * DM * 4;     // 64 MiB
    bf16*  ybf  = (bf16*)w;             w += (size_t)Srows * DM * 2;     // 32 MiB
    bf16*  hbuf = (bf16*)w;             w += (size_t)Srows * DFF * 2;    // 128 MiB
    float* ctx  = (float*)w;            w += (size_t)BS * DC * DM * 4;   // 2 MiB
    float* proj = (float*)w;            w += (size_t)BS * DC * DM * 4;   // 2 MiB
    bf16*  wkb  = (bf16*)w;             w += (size_t)DM * DM * 2;
    bf16*  wob  = (bf16*)w;             w += (size_t)DM * DM * 2;
    bf16*  wf1  = (bf16*)w;             w += (size_t)DFF * DM * 2;
    bf16*  wf2  = (bf16*)w;             w += (size_t)DM * DFF * 2;

    dim3 blk(256);
    // 0. weights -> bf16 (enables pure async-copy tiles in all GEMMs)
    cvt_bf16_kernel<<<dim3((DM * DM) / 256), blk, 0, stream>>>(Wk_w, wkb, DM * DM);
    cvt_bf16_kernel<<<dim3((DM * DM) / 256), blk, 0, stream>>>(out_w, wob, DM * DM);
    cvt_bf16_kernel<<<dim3((DFF * DM) / 256), blk, 0, stream>>>(ff1_w, wf1, DFF * DM);
    cvt_bf16_kernel<<<dim3((DM * DFF) / 256), blk, 0, stream>>>(ff2_w, wf2, DM * DFF);
    // 1. K projection -> bf16 k
    gemm_kernel<float, bf16, 0><<<dim3(DM / 128, Srows / 128), blk, 0, stream>>>(
        src, wkb, Wk_b, kbf, Srows, DM, DM);
    // 2. compressed attention -> ctx [16,128,256]
    attn_kernel<<<dim3(NH, BS), blk, 0, stream>>>(kbf, router, ctx);
    // 3. output projection: [2048,256] = ctx @ out_w^T + out_b
    gemm_kernel<float, float, 0><<<dim3(DM / 128, (BS * DC) / 128), blk, 0, stream>>>(
        ctx, wob, out_b, proj, BS * DC, DM, DM);
    // 4. expand + residual + LN1 -> y (fp32 + bf16)
    expand_ln1_kernel<<<dim3(QL, BS), blk, 0, stream>>>(
        src, proj, pos, m_exp, sc_tf, ln1_g, ln1_b, ybuf, ybf);
    // 5. FFN1 + exact GELU -> bf16 h
    gemm_kernel<bf16, bf16, 1><<<dim3(DFF / 128, Srows / 128), blk, 0, stream>>>(
        ybf, wf1, ff1_b, hbuf, Srows, DFF, DM);
    // 6. FFN2 (bf16 A input)
    gemm_kernel<bf16, float, 0><<<dim3(DM / 128, Srows / 128), blk, 0, stream>>>(
        hbuf, wf2, ff2_b, zbuf, Srows, DM, DFF);
    // 7. residual + LN2 -> out
    add_ln_kernel<<<dim3(Srows), blk, 0, stream>>>(ybuf, zbuf, ln2_g, ln2_b, out);
}
